// AttentionOnDetail_13537736917329
// MI455X (gfx1250) — matrix operand order repaired and checked by reading, not compiled, not run on hardware
//
#include <hip/hip_runtime.h>
#include <hip/hip_bf16.h>

// ---------------------------------------------------------------------------
// AttentionOnDetail for MI455X (gfx1250, wave32, WMMA + TDM)
//
//   GEMM1: h  = relu(x @ W1^T)^2        M=32768 N=256  K=1024   (f32 A -> f16)
//   GEMM2: kv = h @ W2^T                M=32768 N=4096 K=256
//   attn : per-(b,t,head) rmsnorm/rotary/sigmoid + 64-wide scans (wave32)
//   GEMM3: p  = y @ Cw1^T               M=32768 N=256  K=2048
//   GEMM4: p2 = p @ Cw2^T               M=32768 N=2048 K=256
//   gate : out = u * silu(g)            (f32 output)
//
// GEMMs: v_wmma_f32_16x16x32_f16, f32 accumulate. The 64xKC B (weight) panel
// is staged into LDS once per block by the Tensor Data Mover
// (tensor_load_to_lds + s_wait_tensorcnt), with TDM row padding to avoid LDS
// bank conflicts; waves then feed WMMA B fragments from LDS (ds_load_b128).
// ---------------------------------------------------------------------------

typedef __attribute__((ext_vector_type(16))) _Float16 v16h;
typedef __attribute__((ext_vector_type(8)))  _Float16 v8h;
typedef __attribute__((ext_vector_type(8)))  float    v8f;
typedef __attribute__((ext_vector_type(4)))  unsigned int u32x4;
typedef __attribute__((ext_vector_type(8)))  int      i32x8;
typedef __attribute__((ext_vector_type(4)))  int      i32x4;

#define N_EMBD   1024
#define N_HIDDEN 256
#define D_QKV    2048
#define N_HEAD   16
#define HEAD_DIM 64
#define TSEQ     4096
#define BATCH    8
#define MROWS    (BATCH * TSEQ)   // 32768

#define KC    256                 // K-chunk staged in LDS (divides 256/1024/2048)
#define LDSTR (KC + 8)            // halves; 528B row stride (TDM pad 16B/row)

// ---------------------------------------------------------------------- utils
__global__ __launch_bounds__(256) void cvt_f32_to_f16(const float* __restrict__ s,
                                                      _Float16* __restrict__ d, int n) {
  int i = blockIdx.x * 256 + threadIdx.x;
  if (i < n) d[i] = (_Float16)s[i];
}

// ------------------------------------------------------------------ WMMA GEMM
// D[M,N](f16) = epi( A[M,K] @ W[N,K]^T )   (W row-major (N,K), K contiguous)
// block: 256 thr = 8 waves; wave tile 16(M) x 64(N); grid = (M/128, N/64)
// EPI: 0 = identity, 1 = relu^2
template<int EPI, bool A_F32>
__global__ __launch_bounds__(256) void gemm_wmma(const void* __restrict__ Av,
                                                 const _Float16* __restrict__ W,
                                                 _Float16* __restrict__ D,
                                                 int N, int K) {
  __shared__ __attribute__((aligned(128))) _Float16 ldsB[64 * LDSTR];

  const int lane  = threadIdx.x & 31;
  const int wave  = threadIdx.x >> 5;
  const int l15   = lane & 15;
  const bool hiL  = lane >= 16;
  const int mBase = blockIdx.x * 128 + wave * 16;
  const int nBase = blockIdx.y * 64;

  v8f acc[4];
  #pragma unroll
  for (int nt = 0; nt < 4; ++nt) acc[nt] = (v8f){};

  // CDNA5 16-bit A layout: lane<16 -> row=lane, K chunks {0..7},{16..23};
  //                        lane>=16 -> row=lane-16, K chunks {8..15},{24..31}
  const int m    = mBase + l15;
  const int aOff = hiL ? 8  : 0;
  const int bOff = hiL ? 8  : 0;

  const unsigned ldsBase = (unsigned)(uintptr_t)(void*)&ldsB[0];

  for (int kc0 = 0; kc0 < K; kc0 += KC) {
    __syncthreads();   // previous chunk's LDS readers done before overwrite

    if (wave == 0) {
      // Tensor DMA: 64 rows x KC halves of W into LDS, 16B pad per row.
      const unsigned long long ga =
          (unsigned long long)(uintptr_t)(W + (size_t)nBase * K + kc0);
      u32x4 g0;
      g0[0] = 1u;                                          // count=1, user mode
      g0[1] = ldsBase;                                     // lds_addr
      g0[2] = (unsigned)(ga & 0xffffffffu);                // global_addr[31:0]
      g0[3] = (unsigned)((ga >> 32) & 0x01ffffffu)         // global_addr[56:32]
              | (2u << 30);                                // type = 2 (image)
      i32x8 g1;
      g1[0] = (1 << 16)            // data_size = 2 bytes
            | (1 << 20)            // pad_enable
            | (6 << 22)            // pad_interval: every 128 DWORDs (= 1 row)
            | (3 << 25);           // pad_amount: 4 DWORDs (16B) -> 528B stride
      g1[1] = (KC & 0xffff) << 16;                 // tensor_dim0[15:0]
      g1[2] = (KC >> 16) | (64 << 16);             // tensor_dim0[31:16] | tensor_dim1[15:0]
      g1[3] = (KC << 16);                          // tensor_dim1[31:16]=0 | tile_dim0
      g1[4] = 64;                                  // tile_dim1 | tile_dim2=0
      g1[5] = K;                                   // tensor_dim0_stride[31:0]
      g1[6] = 0;                                   // stride hi / dim1_stride
      g1[7] = 0;
      const i32x4 z4 = {0, 0, 0, 0};
      const i32x8 z8 = {0, 0, 0, 0, 0, 0, 0, 0};
      // 6-arg variant (clang-23 / therock-10.0 HIP headers)
      __builtin_amdgcn_tensor_load_to_lds(g0, g1, z4, z4, z8, 0);
    }
    __builtin_amdgcn_s_wait_tensorcnt(0);
    __syncthreads();   // staged B visible to all 8 waves

    for (int kk = 0; kk < KC; kk += 32) {
      v16h a;
      if (A_F32) {
        const float* Af = (const float*)Av + (size_t)m * K + (kc0 + kk + aOff);
        #pragma unroll
        for (int j = 0; j < 8; ++j) {
          a[j]     = (_Float16)Af[j];
          a[j + 8] = (_Float16)Af[16 + j];
        }
      } else {
        const _Float16* Ah = (const _Float16*)Av + (size_t)m * K + (kc0 + kk + aOff);
        v8h alo = *(const v8h*)(Ah);
        v8h ahi = *(const v8h*)(Ah + 16);
        #pragma unroll
        for (int j = 0; j < 8; ++j) { a[j] = alo[j]; a[j + 8] = ahi[j]; }
      }

      #pragma unroll
      for (int nt = 0; nt < 4; ++nt) {
        // B fragment from LDS: lane -> column n = nt*16+l15, K chunks {bOff..+7},{16+bOff..+7}
        const _Float16* Bp = &ldsB[(nt * 16 + l15) * LDSTR + kk + bOff];
        v8h b0 = *(const v8h*)(Bp);
        v8h b1 = *(const v8h*)(Bp + 16);
        v16h b;
        #pragma unroll
        for (int j = 0; j < 8; ++j) { b[j] = b0[j]; b[j + 8] = b1[j]; }
        acc[nt] = __builtin_amdgcn_wmma_f32_16x16x32_f16(
            /*neg_a=*/false, a, /*neg_b=*/false, b,
            /*c_mod=*/(short)0, acc[nt], /*reuse_a=*/false, /*reuse_b=*/false);
      }
    }
  }

  // C/D layout: VGPR j, lane -> row = mBase + j + (lane<16 ? 0 : 8), col = l15
  #pragma unroll
  for (int nt = 0; nt < 4; ++nt) {
    const int col = nBase + nt * 16 + l15;
    #pragma unroll
    for (int j = 0; j < 8; ++j) {
      float c = acc[nt][j];
      if (EPI == 1) { c = fmaxf(c, 0.0f); c = c * c; }
      const int row = mBase + j + (hiL ? 8 : 0);
      D[(size_t)row * N + col] = (_Float16)c;
    }
  }
}

// ------------------------------------------------------- wave32 scan helpers
__device__ __forceinline__ float wave_sum(float x) {
  #pragma unroll
  for (int off = 16; off > 0; off >>= 1) x += __shfl_xor(x, off, 32);
  return x;
}
__device__ __forceinline__ float wave_iscan(float x, int lane) {
  #pragma unroll
  for (int off = 1; off < 32; off <<= 1) {
    float y = __shfl_up(x, off, 32);
    if (lane >= off) x += y;
  }
  return x;
}

// -------------------------------------------- per-head rmsnorm/rotary/scans
// One wave per (b,t,head). Lane i owns d=i and d=i+32 (rotary pair).
__global__ __launch_bounds__(256) void attn_detail(const _Float16* __restrict__ kv,
                                                   _Float16* __restrict__ y) {
  const int lane = threadIdx.x & 31;
  const int wid  = blockIdx.x * 8 + (threadIdx.x >> 5);   // 0 .. MROWS*16-1
  const int head = wid & 15;
  const int rowM = wid >> 4;          // b*T + t
  const int t    = rowM & (TSEQ - 1);

  const _Float16* kvp = kv + (size_t)rowM * (2 * D_QKV);
  float ka = (float)kvp[head * 64 + lane];
  float kb = (float)kvp[head * 64 + 32 + lane];
  float va = (float)kvp[(head + 16) * 64 + lane];
  float vb = (float)kvp[(head + 16) * 64 + 32 + lane];

  // rotary tables: freq[i] = (1/1024)^(i/15) for i<16 else 0 ; theta = t*freq
  const float fr = (lane < 16) ? exp2f(-10.0f * (float)lane * (1.0f / 15.0f)) : 0.0f;
  const float th = (float)t * fr;
  const float c = cosf(th), s = sinf(th);

  // k = rotary(rms_norm(k))
  float ss = wave_sum(ka * ka + kb * kb);
  float r  = rsqrtf(ss * (1.0f / 64.0f) + 1e-6f);
  ka *= r; kb *= r;
  const float ra =  ka * c + kb * s;
  const float rb = -ka * s + kb * c;

  // q = relu(k); k = sigmoid(k)
  float qa = fmaxf(ra, 0.0f), qb = fmaxf(rb, 0.0f);
  const float ska = 1.0f / (1.0f + __expf(-ra));
  const float skb = 1.0f / (1.0f + __expf(-rb));

  // q *= cumsum(k*v,d) / cumsum(k,d)   (d runs 0..63: lanes then +32 half)
  {
    float skv1 = wave_iscan(ska * va, lane);
    float sk1  = wave_iscan(ska, lane);
    float tkv  = __shfl(skv1, 31, 32);
    float tk   = __shfl(sk1, 31, 32);
    float skv2 = tkv + wave_iscan(skb * vb, lane);
    float sk2  = tk  + wave_iscan(skb, lane);
    qa = qa * skv1 / sk1;
    qb = qb * skv2 / sk2;
  }

  // q = rotary(rms_norm(q))
  float ss2 = wave_sum(qa * qa + qb * qb);
  float r2  = rsqrtf(ss2 * (1.0f / 64.0f) + 1e-6f);
  qa *= r2; qb *= r2;
  const float q1 =  qa * c + qb * s;
  const float q2 = -qa * s + qb * c;

  // w = 1 + q*k ; y = cumsum(w*v,d)/cumsum(w,d)
  const float wa = 1.0f + q1 * ska;
  const float wb = 1.0f + q2 * skb;
  float swv1 = wave_iscan(wa * va, lane);
  float sw1  = wave_iscan(wa, lane);
  float twv  = __shfl(swv1, 31, 32);
  float tw   = __shfl(sw1, 31, 32);
  float swv2 = twv + wave_iscan(wb * vb, lane);
  float sw2  = tw  + wave_iscan(wb, lane);

  _Float16* yp = y + (size_t)rowM * D_QKV + head * 64;
  yp[lane]      = (_Float16)(swv1 / sw1);
  yp[lane + 32] = (_Float16)(swv2 / sw2);
}

// ------------------------------------------------------------- final gating
__global__ __launch_bounds__(256) void gate_silu(const _Float16* __restrict__ p2,
                                                 float* __restrict__ out) {
  const size_t idx = (size_t)blockIdx.x * 256 + threadIdx.x;  // rows*1024
  const size_t row = idx >> 10;
  const int    col = (int)(idx & 1023);
  const float u = (float)p2[row * (2 * N_EMBD) + col];
  const float g = (float)p2[row * (2 * N_EMBD) + N_EMBD + col];
  out[idx] = u * (g / (1.0f + __expf(-g)));
}

// ---------------------------------------------------------------------------
extern "C" void kernel_launch(void* const* d_in, const int* in_sizes, int n_in,
                              void* d_out, int out_size, void* d_ws, size_t ws_size,
                              hipStream_t stream) {
  (void)in_sizes; (void)n_in; (void)out_size; (void)ws_size;
  const float* x    = (const float*)d_in[0];
  const float* w1   = (const float*)d_in[1];  // (256, 1024)
  const float* w2   = (const float*)d_in[2];  // (4096, 256)
  const float* cw1  = (const float*)d_in[3];  // (256, 2048)
  const float* cw2  = (const float*)d_in[4];  // (2048, 256)
  float* out        = (float*)d_out;

  // workspace layout (256B-aligned f16 buffers)
  char* ws = (char*)d_ws;
  size_t off = 0;
  auto alloc = [&](size_t bytes) {
    char* p = ws + off;
    off += (bytes + 255) & ~(size_t)255;
    return p;
  };
  _Float16* w1h  = (_Float16*)alloc((size_t)N_HIDDEN * N_EMBD * 2);
  _Float16* w2h  = (_Float16*)alloc((size_t)2 * D_QKV * N_HIDDEN * 2);
  _Float16* cw1h = (_Float16*)alloc((size_t)N_HIDDEN * D_QKV * 2);
  _Float16* cw2h = (_Float16*)alloc((size_t)2 * N_EMBD * N_HIDDEN * 2);
  _Float16* h16  = (_Float16*)alloc((size_t)MROWS * N_HIDDEN * 2);
  _Float16* kv16 = (_Float16*)alloc((size_t)MROWS * 2 * D_QKV * 2);
  _Float16* y16  = (_Float16*)alloc((size_t)MROWS * D_QKV * 2);
  _Float16* p16  = (_Float16*)alloc((size_t)MROWS * N_HIDDEN * 2);
  _Float16* p2h  = (_Float16*)alloc((size_t)MROWS * 2 * N_EMBD * 2);

  // weights -> f16 (tiny; done every call for determinism)
  {
    int n;
    n = N_HIDDEN * N_EMBD;    cvt_f32_to_f16<<<(n + 255) / 256, 256, 0, stream>>>(w1,  w1h,  n);
    n = 2 * D_QKV * N_HIDDEN; cvt_f32_to_f16<<<(n + 255) / 256, 256, 0, stream>>>(w2,  w2h,  n);
    n = N_HIDDEN * D_QKV;     cvt_f32_to_f16<<<(n + 255) / 256, 256, 0, stream>>>(cw1, cw1h, n);
    n = 2 * N_EMBD * N_HIDDEN;cvt_f32_to_f16<<<(n + 255) / 256, 256, 0, stream>>>(cw2, cw2h, n);
  }

  const int mBlocks = MROWS / 128;  // 256

  // GEMM1: h = relu(x @ W1^T)^2      (f32 A, fused epilogue)
  gemm_wmma<1, true><<<dim3(mBlocks, N_HIDDEN / 64), 256, 0, stream>>>(
      (const void*)x, w1h, h16, N_HIDDEN, N_EMBD);

  // GEMM2: kv = h @ W2^T
  gemm_wmma<0, false><<<dim3(mBlocks, (2 * D_QKV) / 64), 256, 0, stream>>>(
      (const void*)h16, w2h, kv16, 2 * D_QKV, N_HIDDEN);

  // per-head rmsnorm / rotary / sigmoid / scans  (one wave32 per head)
  attn_detail<<<(MROWS * N_HEAD) / 8, 256, 0, stream>>>(kv16, y16);

  // GEMM3: p = y @ Cw1^T
  gemm_wmma<0, false><<<dim3(mBlocks, N_HIDDEN / 64), 256, 0, stream>>>(
      (const void*)y16, cw1h, p16, N_HIDDEN, D_QKV);

  // GEMM4: p2 = p @ Cw2^T
  gemm_wmma<0, false><<<dim3(mBlocks, (2 * N_EMBD) / 64), 256, 0, stream>>>(
      (const void*)p16, cw2h, p2h, 2 * N_EMBD, N_HIDDEN);

  // out = u * silu(g)
  gate_silu<<<((size_t)MROWS * N_EMBD) / 256, 256, 0, stream>>>(p2h, out);
}
